// PopulationSNN_55465207660617
// MI455X (gfx1250) — 8-bit matrix operand order repaired and checked by reading, compiled, not run on hardware
//
#include <hip/hip_runtime.h>

typedef __attribute__((ext_vector_type(8))) float v8f;
typedef __attribute__((ext_vector_type(8))) int   v8i;
typedef __attribute__((ext_vector_type(4))) int   v4i;
typedef __attribute__((ext_vector_type(2))) int   v2i;
typedef __attribute__((ext_vector_type(2))) float vf2;

namespace {
constexpr int B  = 2048;
constexpr int N  = 512;
constexpr int T  = 100;
constexpr int H1 = 512;
constexpr int H2 = 256;
constexpr int O  = 5;

constexpr int BT  = 16;
constexpr int NT1 = H1 / 16;
constexpr int NT2 = H2 / 16;
constexpr int KC1 = N  / 64;
constexpr int KC2 = H1 / 64;
constexpr int TC  = 2;

constexpr size_t W1_BYTES = (size_t)KC1 * NT1 * 32 * 32;
constexpr size_t W2_BYTES = (size_t)KC2 * NT2 * 32 * 32;

constexpr unsigned char FP8_ONE = 0x38;
}

__device__ inline v8f wmma_fp8(v8i a, v8i b, v8f c) {
  return __builtin_amdgcn_wmma_f32_16x16x64_fp8_fp8(
      a, b,  (short)0, c,  false,  false);
}

__device__ inline unsigned char f32_to_fp8(float f) {
  unsigned int u  = __float_as_uint(f);
  unsigned int s  = (u >> 24) & 0x80u;
  unsigned int au = u & 0x7fffffffu;
  if (au < 0x3c800000u) {
    float a = __uint_as_float(au);
    int q = (int)(a * 512.0f + 0.5f);
    if (q > 7) q = 7;
    return (unsigned char)(s | (unsigned)q);
  }
  if (au >= 0x43e00000u)
    return (unsigned char)(s | 0x7eu);
  unsigned int m    = au & 0x7fffffu;
  unsigned int keep = m >> 20;
  unsigned int rest = m & 0xfffffu;
  unsigned int rnd  = (rest > 0x80000u || (rest == 0x80000u && (keep & 1u))) ? 1u : 0u;
  unsigned int exp  = au >> 23;
  unsigned int mant = keep + rnd;
  if (mant == 8u) { mant = 0u; exp += 1u; }
  int e8 = (int)exp - 127 + 7;
  if (e8 >= 16) return (unsigned char)(s | 0x7eu);
  return (unsigned char)(s | ((unsigned)e8 << 3) | mant);
}

__device__ inline v8i load_frag_a8(const unsigned char* base, int stride, int k0,
                                   int l16, int hi) {
  const unsigned char* row = base + l16 * stride + k0 + (hi ? 8 : 0);
  v2i d0 = *(const v2i*)(row);
  v2i d1 = *(const v2i*)(row + 16);
  v2i d2 = *(const v2i*)(row + 32);
  v2i d3 = *(const v2i*)(row + 48);
  v4i lo = __builtin_shufflevector(d0, d1, 0, 1, 2, 3);
  v4i hh = __builtin_shufflevector(d2, d3, 0, 1, 2, 3);
  return __builtin_shufflevector(lo, hh, 0, 1, 2, 3, 4, 5, 6, 7);
}

__device__ inline v8i load_frag_b8(const unsigned char* packed, int fragIdx,
                                   int lane) {
  const unsigned char* p = packed + ((size_t)fragIdx * 32 + lane) * 32;
  v4i lo = *(const v4i*)(p);
  v4i hh = *(const v4i*)(p + 16);
  return __builtin_shufflevector(lo, hh, 0, 1, 2, 3, 4, 5, 6, 7);
}

__global__ void snn_pack_weights_fp8(const float* __restrict__ W1,
                                     const float* __restrict__ W2,
                                     unsigned char* __restrict__ W1p,
                                     unsigned char* __restrict__ W2p) {
  size_t idx = (size_t)blockIdx.x * blockDim.x + threadIdx.x;
  if (idx < W1_BYTES) {
    int fragIdx = (int)(idx >> 10);
    int r    = (int)(idx & 1023);
    int lane = r >> 5;
    int bb   = r & 31;
    int kc = fragIdx / NT1, jt = fragIdx % NT1;
    int l16 = lane & 15, hi = lane >> 4;
    int grp = bb >> 4, j = bb & 15;
    int k = kc * 64 + grp * 32 + (j >> 3) * 16 + hi * 8 + (j & 7);
    int n = jt * 16 + l16;
    W1p[idx] = f32_to_fp8(W1[(size_t)n * N + k]);
  } else if (idx < W1_BYTES + W2_BYTES) {
    size_t i2 = idx - W1_BYTES;
    int fragIdx = (int)(i2 >> 10);
    int r    = (int)(i2 & 1023);
    int lane = r >> 5;
    int bb   = r & 31;
    int kc = fragIdx / NT2, jt = fragIdx % NT2;
    int l16 = lane & 15, hi = lane >> 4;
    int grp = bb >> 4, j = bb & 15;
    int k = kc * 64 + grp * 32 + (j >> 3) * 16 + hi * 8 + (j & 7);
    int n = jt * 16 + l16;
    W2p[i2] = f32_to_fp8(W2[(size_t)n * H1 + k]);
  }
}

__global__ __launch_bounds__(256)
void snn_persistent_kernel(const float* __restrict__ x,
                           const float* __restrict__ b1,
                           const float* __restrict__ b2,
                           const float* __restrict__ Wo,
                           const float* __restrict__ bo,
                           const unsigned char* __restrict__ W1p,
                           const unsigned char* __restrict__ W2p,
                           float* __restrict__ out) {
  __shared__ alignas(16) unsigned char w1lds[W1_BYTES];
  __shared__ alignas(16) unsigned char xh[TC][BT][N];
  __shared__ alignas(16) unsigned char s1h[BT][H1];
  __shared__ alignas(16) unsigned char s2h[BT][H2];

  const int tid  = threadIdx.x;
  const int lane = tid & 31;
  const int wv   = tid >> 5;
  const int l16  = lane & 15;
  const int hi   = lane >> 4;
  const int b0   = blockIdx.x * BT;

  {
    constexpr int ITERS = (int)(W1_BYTES / (256 * 16));
    for (int it = 0; it < ITERS; ++it) {
      size_t off = (size_t)it * 4096 + (size_t)tid * 16;
      unsigned lofs = (unsigned)(size_t)&w1lds[off];
      unsigned long long ga = (unsigned long long)(W1p + off);
      asm volatile("global_load_async_to_lds_b128 %0, %1, off"
                   :: "v"(lofs), "v"(ga) : "memory");
    }
    asm volatile("s_wait_asynccnt 0" ::: "memory");
  }

  v8f v1s[4], v2s[2];
  for (int nt = 0; nt < 4; ++nt)
    for (int i = 0; i < 8; ++i) v1s[nt][i] = 0.0f;
  for (int nt = 0; nt < 2; ++nt)
    for (int i = 0; i < 8; ++i) v2s[nt][i] = 0.0f;

  float bias1[4], bias2[2];
  for (int nt = 0; nt < 4; ++nt) bias1[nt] = b1[(wv * 4 + nt) * 16 + l16];
  for (int nt = 0; nt < 2; ++nt) bias2[nt] = b2[(wv * 2 + nt) * 16 + l16];

  const int m3 = tid / O;
  const int o3 = tid % O;
  float vo = 0.0f, so_sum = 0.0f;
  const float bo3 = (tid < BT * O) ? bo[o3] : 0.0f;

  __syncthreads();

  for (int t0 = 0; t0 < T; t0 += TC) {
    __syncthreads();
    for (int idx = tid; idx < BT * N; idx += 256) {
      int m = idx >> 9;
      int n = idx & (N - 1);
      const float* px = x + ((size_t)(b0 + m) * N + n) * T + t0;
      vf2 xv = *(const vf2*)px;
      xh[0][m][n] = f32_to_fp8(xv.x);
      xh[1][m][n] = f32_to_fp8(xv.y);
      if (t0 + TC < T) __builtin_prefetch(px + TC, 0, 0);
    }
    __syncthreads();

    for (int ti = 0; ti < TC; ++ti) {
      v8f acc[4];
      for (int nt = 0; nt < 4; ++nt)
        for (int i = 0; i < 8; ++i) acc[nt][i] = 0.0f;
      #pragma unroll
      for (int kc = 0; kc < KC1; ++kc) {
        v8i a = load_frag_a8(&xh[ti][0][0], N, kc * 64, l16, hi);
        #pragma unroll
        for (int nt = 0; nt < 4; ++nt) {
          v8i bfrag = load_frag_b8(w1lds, kc * NT1 + (wv * 4 + nt), lane);
          acc[nt] = wmma_fp8(a, bfrag, acc[nt]);
        }
      }
      #pragma unroll
      for (int nt = 0; nt < 4; ++nt) {
        int j = (wv * 4 + nt) * 16 + l16;
        #pragma unroll
        for (int i = 0; i < 8; ++i) {
          float cur = acc[nt][i] + bias1[nt];
          float v = v1s[nt][i];
          v += (cur - v) * 0.5f;
          bool  sp = (v >= 1.0f);
          v1s[nt][i] = sp ? 0.0f : v;
          s1h[i + hi * 8][j] = sp ? FP8_ONE : (unsigned char)0;
        }
      }
      __syncthreads();

      v8f acc2[2];
      for (int nt = 0; nt < 2; ++nt)
        for (int i = 0; i < 8; ++i) acc2[nt][i] = 0.0f;
      #pragma unroll
      for (int kc = 0; kc < KC2; ++kc) {
        v8i a = load_frag_a8(&s1h[0][0], H1, kc * 64, l16, hi);
        #pragma unroll
        for (int nt = 0; nt < 2; ++nt) {
          v8i bfrag = load_frag_b8(W2p, kc * NT2 + (wv * 2 + nt), lane);
          acc2[nt] = wmma_fp8(a, bfrag, acc2[nt]);
        }
      }
      #pragma unroll
      for (int nt = 0; nt < 2; ++nt) {
        int j = (wv * 2 + nt) * 16 + l16;
        #pragma unroll
        for (int i = 0; i < 8; ++i) {
          float cur = acc2[nt][i] + bias2[nt];
          float v = v2s[nt][i];
          v += (cur - v) * 0.5f;
          bool  sp = (v >= 1.0f);
          v2s[nt][i] = sp ? 0.0f : v;
          s2h[i + hi * 8][j] = sp ? FP8_ONE : (unsigned char)0;
        }
      }
      __syncthreads();

      if (tid < BT * O) {
        float dot = bo3;
        for (int k = 0; k < H2; k += 4) {
          unsigned u = *(const unsigned*)&s2h[m3][k];
          #pragma unroll
          for (int j = 0; j < 4; ++j) {
            float sj = ((u >> (8 * j)) & 0xffu) ? 1.0f : 0.0f;
            dot = fmaf(sj, Wo[o3 * H2 + k + j], dot);
          }
        }
        float v = vo;
        v += (dot - v) * 0.5f;
        bool sp = (v >= 1.0f);
        vo = sp ? 0.0f : v;
        so_sum += sp ? 1.0f : 0.0f;
      }
    }
  }

  if (tid < BT * O) {
    out[(size_t)(b0 + m3) * O + o3] = so_sum * (1.0f / T);
  }
}

extern "C" void kernel_launch(void* const* d_in, const int* in_sizes, int n_in,
                              void* d_out, int out_size, void* d_ws, size_t ws_size,
                              hipStream_t stream) {
  const float* x  = (const float*)d_in[0];
  const float* W1 = (const float*)d_in[1];
  const float* b1 = (const float*)d_in[2];
  const float* W2 = (const float*)d_in[3];
  const float* b2 = (const float*)d_in[4];
  const float* Wo = (const float*)d_in[5];
  const float* bo = (const float*)d_in[6];
  float* out = (float*)d_out;

  unsigned char* W1p = (unsigned char*)d_ws;
  unsigned char* W2p = W1p + W1_BYTES;

  {
    size_t total = W1_BYTES + W2_BYTES;
    int threads = 256;
    int blocks = (int)((total + threads - 1) / threads);
    snn_pack_weights_fp8<<<blocks, threads, 0, stream>>>(W1, W2, W1p, W2p);
  }
  snn_persistent_kernel<<<B / BT, 256, 0, stream>>>(x, b1, b2, Wo, bo,
                                                    W1p, W2p, out);
}
